// BiLSTM_Atten_CRF_51204600103387
// MI455X (gfx1250) — matrix operand order repaired and checked by reading, not compiled, not run on hardware
//
#include <hip/hip_runtime.h>
#include <stdint.h>

typedef unsigned short u16;
typedef __attribute__((ext_vector_type(16))) __bf16 v16bf;
typedef __attribute__((ext_vector_type(8)))  float  v8f;

#define WMMA_BF16(A_, B_, C_) \
  __builtin_amdgcn_wmma_f32_16x16x32_bf16(false, (A_), false, (B_), (short)0, (C_), false, false)

struct Uint4x2 { uint4 lo, hi; };
struct U16x16  { u16 u[16];   };

__device__ __forceinline__ u16 f32_to_bf16(float f) {
  uint32_t x = __builtin_bit_cast(uint32_t, f);
  x += 0x7FFFu + ((x >> 16) & 1u);              // round-to-nearest-even
  return (u16)(x >> 16);
}

__device__ __forceinline__ float sigm(float x) { return 1.0f / (1.0f + __expf(-x)); }

// A fragment, 16x32 bf16 (M x K), row-major source with leading dim lda.
// ISA layout: lane L: m = L%16, kb = (L/16)*8; elems 0..7 -> K=kb+e, 8..15 -> K=kb+16+(e-8).
__device__ __forceinline__ v16bf load_a_frag(const u16* A, int lda, int m0, int k0, int lane) {
  const int m  = m0 + (lane & 15);
  const int kb = k0 + ((lane >> 4) << 3);
  const u16* p = A + (size_t)m * lda + kb;
  Uint4x2 t;
  t.lo = *(const uint4*)(p);
  t.hi = *(const uint4*)(p + 16);
  return __builtin_bit_cast(v16bf, t);
}

// B fragment, 32x16 bf16 (K x N), computed from W stored [N,K] row-major (so we do A*W^T).
// ISA layout: lane L: n = L%16, kb = (L/16)*8; elems 0..7 -> K=kb+e, 8..15 -> K=kb+16+(e-8).
__device__ __forceinline__ v16bf load_b_frag(const u16* W, int ldw, int n0, int k0, int lane) {
  const int n  = n0 + (lane & 15);
  const int ks = k0 + ((lane >> 4) << 3);
  const u16* p = W + (size_t)n * ldw + ks;
  Uint4x2 t;
  t.lo = *(const uint4*)(p);
  t.hi = *(const uint4*)(p + 16);
  return __builtin_bit_cast(v16bf, t);
}

// ---------------------------------------------------------------- converters
__global__ void f32_to_bf16_kernel(const float* __restrict__ in, u16* __restrict__ out, int n) {
  for (int i = blockIdx.x * blockDim.x + threadIdx.x; i < n; i += gridDim.x * blockDim.x)
    out[i] = f32_to_bf16(in[i]);
}

// e[(s*32+b), :] = bf16(emb[x[b,s], :])   (time-major activation layout [S,B,E])
__global__ void embed_gather(const int* __restrict__ x, const float* __restrict__ emb,
                             u16* __restrict__ e) {
  const int r = blockIdx.x;            // 0..16383
  const int s = r >> 5, b = r & 31;
  const int tok = x[b * 512 + s];
  const float* src = emb + (size_t)tok * 512;
  u16* dst = e + (size_t)r * 512;
  for (int i = threadIdx.x; i < 512; i += blockDim.x) dst[i] = f32_to_bf16(src[i]);
}

// ------------------------------------------------- generic WMMA GEMM (A*W^T)
// A[M,K] bf16 row-major, W[N,K] bf16 row-major, out f32 or bf16, optional bias[N].
// Block tile 128x64, 8 waves, each wave 32x32 (2x2 WMMA tiles). Weights/activations
// are L2-resident (192MB L2), so fragments stream straight from global with prefetch.
template <bool OUT_BF16>
__global__ __launch_bounds__(256) void gemm_bf16(
    const u16* __restrict__ A, const u16* __restrict__ W, int M, int N, int K,
    float* __restrict__ Cf, u16* __restrict__ Cb, const float* __restrict__ bias) {
  const int lane = threadIdx.x & 31, w = threadIdx.x >> 5;
  const int m0 = blockIdx.x * 128 + (w & 3) * 32;
  const int n0 = blockIdx.y * 64 + (w >> 2) * 32;
  v8f acc00 = {}, acc01 = {}, acc10 = {}, acc11 = {};
  for (int k0 = 0; k0 < K; k0 += 32) {
    if (k0 + 32 < K) {
      __builtin_prefetch(A + (size_t)(m0 + (lane & 15)) * K + k0 + 32, 0, 1);
      __builtin_prefetch(W + (size_t)(n0 + (lane & 15)) * K + k0 + 32, 0, 1);
    }
    v16bf a0 = load_a_frag(A, K, m0, k0, lane);
    v16bf a1 = load_a_frag(A, K, m0 + 16, k0, lane);
    v16bf b0 = load_b_frag(W, K, n0, k0, lane);
    v16bf b1 = load_b_frag(W, K, n0 + 16, k0, lane);
    acc00 = WMMA_BF16(a0, b0, acc00);
    acc01 = WMMA_BF16(a0, b1, acc01);
    acc10 = WMMA_BF16(a1, b0, acc10);
    acc11 = WMMA_BF16(a1, b1, acc11);
  }
  v8f accs[4] = {acc00, acc01, acc10, acc11};
  const int nn = lane & 15, mb = (lane >> 4) << 3;   // C/D: vgpr v -> M = mb+v, N = nn
#pragma unroll
  for (int mt = 0; mt < 2; ++mt)
#pragma unroll
    for (int nt = 0; nt < 2; ++nt) {
      v8f a = accs[mt * 2 + nt];
      const int n = n0 + nt * 16 + nn;
      const float bv = bias ? bias[n] : 0.0f;
#pragma unroll
      for (int v = 0; v < 8; ++v) {
        const size_t idx = (size_t)(m0 + mt * 16 + mb + v) * N + n;
        const float val = a[v] + bv;
        if constexpr (OUT_BF16) Cb[idx] = f32_to_bf16(val);
        else                    Cf[idx] = val;
      }
    }
}

// ------------------------------------------------------- persistent BiLSTM
// Grid = 16 WGs (dir = bx/8, hidden-group = bx%8). WG owns 32 hidden units j:
// Whh rows {j, 256+j, 512+j, 768+j} -> 128x256 bf16 slice resident in LDS (64KB).
// Each step: z = h_prev @ Whh_slice^T (WMMA) + zin[t] + bias -> gates -> c,h.
// h state ping-pongs via L2; grid sync = monotonic agent-scope atomic + s_sleep.
__global__ __launch_bounds__(256) void lstm_layer(
    const float* __restrict__ zin,   // [2][S*32][1024] input-proj pre-activations
    const float* __restrict__ bih,   // layer slice [2][1024]
    const float* __restrict__ bhh,   // layer slice [2][1024]
    const u16*  __restrict__ whh,    // layer slice [2][1024][256] bf16
    u16* __restrict__ hout,          // [S*32][512] bf16 (fwd cols 0..255, bwd 256..511)
    u16* __restrict__ hstate,        // [2 dir][2 ping][32*256] bf16
    unsigned* __restrict__ bar) {
  extern __shared__ char smem[];
  u16*   wlds = (u16*)smem;                         // 128*256 u16  = 64 KB
  float* zlds = (float*)(smem + 65536);             // 32*128 f32   = 16 KB
  float* clds = (float*)(smem + 65536 + 16384);     // 32*32  f32   =  4 KB
  const int tid = threadIdx.x, lane = tid & 31, w = tid >> 5;
  const int dir = blockIdx.x >> 3, grp = blockIdx.x & 7;

  // stage Whh slice into LDS, rows ordered [i(32) | f(32) | g(32) | o(32)]
  const u16* wsrc = whh + (size_t)dir * 262144;
  for (int i = tid; i < 4096; i += 256) {           // 4096 chunks of 8 u16
    const int lr = i >> 5, cc = (i & 31) << 3;
    const int gate = lr >> 5, jl = lr & 31;
    const int grow = gate * 256 + grp * 32 + jl;
    *(uint4*)(wlds + lr * 256 + cc) = *(const uint4*)(wsrc + (size_t)grow * 256 + cc);
  }
  __syncthreads();

  const float* zin_d = zin + (size_t)dir * (512u * 32u * 1024u);
  const float* bi = bih + dir * 1024;
  const float* bh = bhh + dir * 1024;

  for (int t = 0; t < 512; ++t) {
    const int tr = dir ? (511 - t) : t;
    const int pin = t & 1, pout = pin ^ 1;
    v8f acc0 = {}, acc1 = {};
    if (t > 0) {                                    // h_prev == 0 at t==0
      const u16* hp = hstate + (size_t)(dir * 2 + pin) * 8192;
#pragma unroll
      for (int kc = 0; kc < 8; ++kc) {
        v16bf a0 = load_a_frag(hp, 256, 0, kc * 32, lane);
        v16bf a1 = load_a_frag(hp, 256, 16, kc * 32, lane);
        v16bf bf = load_b_frag(wlds, 256, w * 16, kc * 32, lane);
        acc0 = WMMA_BF16(a0, bf, acc0);
        acc1 = WMMA_BF16(a1, bf, acc1);
      }
    }
    {
      const int n = w * 16 + (lane & 15), mb = (lane >> 4) << 3;
#pragma unroll
      for (int v = 0; v < 8; ++v) {
        zlds[(mb + v) * 128 + n]      = acc0[v];
        zlds[(16 + mb + v) * 128 + n] = acc1[v];
      }
    }
    __syncthreads();
#pragma unroll
    for (int q = 0; q < 4; ++q) {
      const int idx = tid + q * 256;                // (b,j): 32x32 pairs
      const int b = idx >> 5, j = idx & 31;
      const int jg = grp * 32 + j;
      const size_t r = (size_t)tr * 32 + b;
      const float* zrow = zin_d + r * 1024;
      const float zi = zlds[b * 128 + j]      + zrow[jg]       + bi[jg]       + bh[jg];
      const float zf = zlds[b * 128 + 32 + j] + zrow[256 + jg] + bi[256 + jg] + bh[256 + jg];
      const float zg = zlds[b * 128 + 64 + j] + zrow[512 + jg] + bi[512 + jg] + bh[512 + jg];
      const float zo = zlds[b * 128 + 96 + j] + zrow[768 + jg] + bi[768 + jg] + bh[768 + jg];
      float c = t ? clds[idx] : 0.0f;
      c = sigm(zf) * c + sigm(zi) * tanhf(zg);
      clds[idx] = c;
      const float h = sigm(zo) * tanhf(c);
      const u16 hb = f32_to_bf16(h);
      hstate[(size_t)(dir * 2 + pout) * 8192 + b * 256 + jg] = hb;
      hout[r * 512 + dir * 256 + jg] = hb;
    }
    __syncthreads();
    if (tid == 0) {                                 // 16-WG grid step barrier
      __hip_atomic_fetch_add(bar, 1u, __ATOMIC_RELEASE, __HIP_MEMORY_SCOPE_AGENT);
      const unsigned target = (unsigned)(t + 1) * 16u;
      while (__hip_atomic_load(bar, __ATOMIC_ACQUIRE, __HIP_MEMORY_SCOPE_AGENT) < target)
        __builtin_amdgcn_s_sleep(2);
    }
    __syncthreads();
  }
}

// ---------------------------------------------------------------- attention
// Reference bug preserved: attention axis is B (=32), "batch" axis is S, 2 heads.
// One WG (4 waves) per (s, head): scores 32x32 (K=256) -> softmax -> out 32x256 (K=32).
__global__ __launch_bounds__(128) void mha_kernel(const u16* __restrict__ qkv,
                                                  u16* __restrict__ attn_out) {
  __shared__ __align__(16) float sc[32 * 32];
  __shared__ __align__(16) u16  att[32 * 32];
  const int tid = threadIdx.x, lane = tid & 31, w = tid >> 5;
  const int s = blockIdx.x >> 1, head = blockIdx.x & 1;
  const u16* base = qkv + (size_t)s * 32 * 1536 + head * 256;
  const u16* Q = base;            // rows l (=b), stride 1536
  const u16* Kk = base + 512;
  const u16* V = base + 1024;

  // scores: wave w computes 16x16 tile (mt, nt)
  const int mt = w & 1, nt = w >> 1;
  v8f acc = {};
#pragma unroll
  for (int kc = 0; kc < 8; ++kc) {
    v16bf a = load_a_frag(Q, 1536, mt * 16, kc * 32, lane);
    v16bf b = load_b_frag(Kk, 1536, nt * 16, kc * 32, lane);
    acc = WMMA_BF16(a, b, acc);
  }
  {
    const int n = nt * 16 + (lane & 15), mb = mt * 16 + ((lane >> 4) << 3);
#pragma unroll
    for (int v = 0; v < 8; ++v) sc[(mb + v) * 32 + n] = acc[v] * 0.0625f;  // 1/sqrt(256)
  }
  __syncthreads();
  if (tid < 32) {                                   // row-wise softmax over m
    float mx = -1e30f;
    for (int m = 0; m < 32; ++m) mx = fmaxf(mx, sc[tid * 32 + m]);
    float sum = 0.0f;
    for (int m = 0; m < 32; ++m) sum += __expf(sc[tid * 32 + m] - mx);
    const float inv = 1.0f / sum;
    for (int m = 0; m < 32; ++m) att[tid * 32 + m] = f32_to_bf16(__expf(sc[tid * 32 + m] - mx) * inv);
  }
  __syncthreads();

  // out = att(32x32) @ V(32x256); wave w owns 64 output cols; K=32 = one WMMA
  v16bf a0 = load_a_frag(att, 32, 0, 0, lane);
  v16bf a1 = load_a_frag(att, 32, 16, 0, lane);
#pragma unroll
  for (int ntile = 0; ntile < 4; ++ntile) {
    const int n0 = w * 64 + ntile * 16;
    const int d = n0 + (lane & 15);
    const int ks = (lane >> 4) << 3;
    U16x16 bt;                                      // B(k,n) = V[k, d]: column gather
#pragma unroll
    for (int e = 0; e < 16; ++e) bt.u[e] = V[(size_t)(ks + e + (e >= 8 ? 8 : 0)) * 1536 + d];
    v16bf bf = __builtin_bit_cast(v16bf, bt);
    v8f o0 = {}, o1 = {};
    o0 = WMMA_BF16(a0, bf, o0);
    o1 = WMMA_BF16(a1, bf, o1);
    const int n = head * 256 + d;
    const int mb = (lane >> 4) << 3;
#pragma unroll
    for (int v = 0; v < 8; ++v) {
      attn_out[((size_t)s * 32 + mb + v) * 512 + n]      = f32_to_bf16(o0[v]);
      attn_out[((size_t)s * 32 + 16 + mb + v) * 512 + n] = f32_to_bf16(o1[v]);
    }
  }
}

// ------------------------------------------- FC (N=16) with [S,B]->[B,S] remap
__global__ __launch_bounds__(256) void fc_kernel(const u16* __restrict__ X,
                                                 const u16* __restrict__ Wfc,
                                                 const float* __restrict__ bias,
                                                 float* __restrict__ out) {
  const int lane = threadIdx.x & 31, w = threadIdx.x >> 5;
  const int m0 = blockIdx.x * 128 + w * 16;
  v8f acc = {};
  for (int k0 = 0; k0 < 512; k0 += 32) {
    v16bf a = load_a_frag(X, 512, m0, k0, lane);
    v16bf b = load_b_frag(Wfc, 512, 0, k0, lane);
    acc = WMMA_BF16(a, b, acc);
  }
  const int n = lane & 15, mb = (lane >> 4) << 3;
  const float bv = bias[n];
#pragma unroll
  for (int v = 0; v < 8; ++v) {
    const int r = m0 + mb + v;                      // r = s*32 + b
    const int sI = r >> 5, bI = r & 31;
    out[((size_t)bI * 512 + sI) * 16 + n] = acc[v] + bv;
  }
}

// ------------------------------------------------------------------ CRF NLL
__global__ __launch_bounds__(512) void crf_kernel(
    const float* __restrict__ logits,  // d_out, [B,S,C] f32
    const int* __restrict__ labels, const float* __restrict__ cstart,
    const float* __restrict__ cend, const float* __restrict__ ctrans,
    float* __restrict__ nll_out) {
  __shared__ float trans[256], al[512], startv[16], endv[16], partial[32];
  const int tid = threadIdx.x;
  if (tid < 256) trans[tid] = ctrans[tid];
  if (tid < 16) { startv[tid] = cstart[tid]; endv[tid] = cend[tid]; }
  __syncthreads();
  const int b = tid >> 4, c = tid & 15;
  float a = startv[c] + logits[((size_t)b * 512) * 16 + c];
  for (int t = 1; t < 512; ++t) {
    al[tid] = a;
    __syncthreads();
    float mx = -1e30f;
#pragma unroll
    for (int p = 0; p < 16; ++p) mx = fmaxf(mx, al[b * 16 + p] + trans[p * 16 + c]);
    float sum = 0.0f;
#pragma unroll
    for (int p = 0; p < 16; ++p) sum += __expf(al[b * 16 + p] + trans[p * 16 + c] - mx);
    a = mx + __logf(sum) + logits[((size_t)b * 512 + t) * 16 + c];
    __syncthreads();
  }
  al[tid] = a + endv[c];
  __syncthreads();
  if (c == 0) {
    float mx = -1e30f;
    for (int p = 0; p < 16; ++p) mx = fmaxf(mx, al[b * 16 + p]);
    float sum = 0.0f;
    for (int p = 0; p < 16; ++p) sum += __expf(al[b * 16 + p] - mx);
    const float logZ = mx + __logf(sum);
    const int* lb = labels + b * 512;
    int prev = lb[0];
    float sc2 = startv[prev] + logits[((size_t)b * 512) * 16 + prev];
    for (int t = 1; t < 512; ++t) {
      const int cur = lb[t];
      sc2 += trans[prev * 16 + cur] + logits[((size_t)b * 512 + t) * 16 + cur];
      prev = cur;
    }
    sc2 += endv[prev];
    partial[b] = sc2 - logZ;
  }
  __syncthreads();
  if (tid == 0) {
    float s2 = 0.0f;
    for (int i = 0; i < 32; ++i) s2 += partial[i];
    nll_out[0] = -s2;
  }
}

// ------------------------------------------------------------------- launch
extern "C" void kernel_launch(void* const* d_in, const int* in_sizes, int n_in,
                              void* d_out, int out_size, void* d_ws, size_t ws_size,
                              hipStream_t stream) {
  (void)in_sizes; (void)n_in; (void)out_size; (void)ws_size;
  const int*   xi   = (const int*)d_in[0];
  const int*   lbls = (const int*)d_in[1];
  const float* emb  = (const float*)d_in[2];
  const float* wih  = (const float*)d_in[3];   // [2,2,1024,512]
  const float* whh  = (const float*)d_in[4];   // [2,2,1024,256]
  const float* bih  = (const float*)d_in[5];   // [2,2,1024]
  const float* bhh  = (const float*)d_in[6];
  const float* inpw = (const float*)d_in[7];   // [1536,512]
  const float* inpb = (const float*)d_in[8];
  const float* outw = (const float*)d_in[9];   // [512,512]
  const float* outb = (const float*)d_in[10];
  const float* fcw  = (const float*)d_in[11];  // [16,512]
  const float* fcb  = (const float*)d_in[12];
  const float* cs   = (const float*)d_in[13];
  const float* ce   = (const float*)d_in[14];
  const float* ct   = (const float*)d_in[15];
  float* out = (float*)d_out;

  char* ws = (char*)d_ws;
  const size_t MB = 1ull << 20;
  // bf16 weight arena (first 16 MB)
  u16* w_wih  = (u16*)ws;                  // 2,097,152
  u16* w_whh  = w_wih + 2097152;           // 1,048,576
  u16* w_inp  = w_whh + 1048576;           //   786,432
  u16* w_outp = w_inp + 786432;            //   262,144
  u16* w_fc   = w_outp + 262144;           //     8,192
  u16*   e_bf = (u16*)(ws + 16 * MB);      // [S*B,512] bf16; later reused as attn_out
  u16*   h0   = (u16*)(ws + 32 * MB);      // layer0 out; later reused as attn_fin
  u16*   h1   = (u16*)(ws + 48 * MB);      // layer1 out
  float* zin  = (float*)(ws + 64 * MB);    // [2][S*B][1024] f32 (128 MB)
  u16*   qkv  = (u16*)(ws + 64 * MB);      // overlay (zin dead after LSTM)
  u16* attn_out = e_bf;
  u16* attn_fin = h0;
  u16* hstate = (u16*)(ws + 192 * MB);     // 2*2*8192 bf16
  unsigned* bars = (unsigned*)(ws + 192 * MB + 65536);

  hipMemsetAsync(bars, 0, 64, stream);     // reset per-layer step barriers

  f32_to_bf16_kernel<<<512, 256, 0, stream>>>(wih, w_wih, 2097152);
  f32_to_bf16_kernel<<<512, 256, 0, stream>>>(whh, w_whh, 1048576);
  f32_to_bf16_kernel<<<512, 256, 0, stream>>>(inpw, w_inp, 786432);
  f32_to_bf16_kernel<<<256, 256, 0, stream>>>(outw, w_outp, 262144);
  f32_to_bf16_kernel<<<32, 256, 0, stream>>>(fcw, w_fc, 8192);
  embed_gather<<<16384, 256, 0, stream>>>(xi, emb, e_bf);

  for (int l = 0; l < 2; ++l) {
    const u16* Ain = (l == 0) ? e_bf : h0;
    u16* Hout = (l == 0) ? h0 : h1;
    for (int d = 0; d < 2; ++d) {
      gemm_bf16<false><<<dim3(128, 16), 256, 0, stream>>>(
          Ain, w_wih + (size_t)(l * 2 + d) * 524288, 16384, 1024, 512,
          zin + (size_t)d * 16777216, nullptr, nullptr);
    }
    lstm_layer<<<16, 256, 86016, stream>>>(
        zin, bih + l * 2048, bhh + l * 2048, w_whh + (size_t)l * 524288,
        Hout, hstate, bars + l);
  }

  gemm_bf16<true><<<dim3(128, 24), 256, 0, stream>>>(h1, w_inp, 16384, 1536, 512,
                                                     nullptr, qkv, inpb);
  mha_kernel<<<1024, 128, 0, stream>>>(qkv, attn_out);
  gemm_bf16<true><<<dim3(128, 8), 256, 0, stream>>>(attn_out, w_outp, 16384, 512, 512,
                                                    nullptr, attn_fin, outb);
  fc_kernel<<<128, 256, 0, stream>>>(attn_fin, w_fc, fcb, out);
  crf_kernel<<<1, 512, 0, stream>>>(out, lbls, cs, ce, ct, out + 262144);
}
